// GUIDEModel_22016002359422
// MI455X (gfx1250) — matrix operand order repaired and checked by reading, not compiled, not run on hardware
//
#include <hip/hip_runtime.h>
#include <hip/hip_bf16.h>
#include <math.h>

typedef _Float16 v16h __attribute__((ext_vector_type(16)));
typedef float    v8f  __attribute__((ext_vector_type(8)));

// ---------------------------------------------------------------------------
// Utility kernels
// ---------------------------------------------------------------------------
__global__ __launch_bounds__(256) void zero_f32_kernel(float* __restrict__ p, long long n) {
  long long id = (long long)blockIdx.x * blockDim.x + threadIdx.x;
  long long stride = (long long)gridDim.x * blockDim.x;
  for (; id < n; id += stride) p[id] = 0.0f;
}

__global__ __launch_bounds__(256) void fill_u32_kernel(unsigned* __restrict__ p, unsigned v, long long n) {
  long long id = (long long)blockIdx.x * blockDim.x + threadIdx.x;
  long long stride = (long long)gridDim.x * blockDim.x;
  for (; id < n; id += stride) p[id] = v;
}

__global__ __launch_bounds__(256) void degree_kernel(const int* __restrict__ src, const int* __restrict__ dst,
                                                     float* __restrict__ degOut, float* __restrict__ degIn, int E) {
  int e = blockIdx.x * blockDim.x + threadIdx.x;
  if (e >= E) return;
  atomicAdd(&degOut[src[e]], 1.0f);
  atomicAdd(&degIn[dst[e]], 1.0f);
}

__global__ __launch_bounds__(256) void rsqrt_clip_kernel(float* __restrict__ d, int n) {
  int i = blockIdx.x * blockDim.x + threadIdx.x;
  if (i >= n) return;
  float v = d[i];
  v = (v < 1.0f) ? 1.0f : v;
  d[i] = rsqrtf(v);
}

// Xh[n,k] = (f16)(X[n,k] * rsOut[n])
__global__ __launch_bounds__(256) void convert_scale_f16_kernel(const float* __restrict__ X,
                                                                const float* __restrict__ rs,
                                                                _Float16* __restrict__ Xh,
                                                                int N, int K) {
  long long id = (long long)blockIdx.x * blockDim.x + threadIdx.x;
  long long total = (long long)N * K;
  if (id >= total) return;
  int n = (int)(id / K);
  Xh[id] = (_Float16)(X[id] * rs[n]);
}

// Wt[o,k] = (f16) W[k,o]   (W is [K,O] row-major)
__global__ __launch_bounds__(256) void transpose_w_f16_kernel(const float* __restrict__ W,
                                                              _Float16* __restrict__ Wt,
                                                              int K, int O) {
  int id = blockIdx.x * blockDim.x + threadIdx.x;
  if (id >= K * O) return;
  int k = id / O;
  int o = id % O;
  Wt[(size_t)o * K + k] = (_Float16)W[id];
}

// ---------------------------------------------------------------------------
// WMMA GEMM:  H[M,Nc] = Xh[M,K] * Wt[Nc,K]^T   (f16 inputs, f32 accumulate)
// N-register-blocked: one wave computes a 16x64 strip (4 x 16x16 tiles),
// reusing one A fragment against 4 B fragments per K-step (4 WMMAs/step).
// A fragment (16x32, 16-bit): lane&15 = M row; VGPR j holds
//   K = (lane>>4)*16 + j*2, +1   (same order as B)
// B fragment (32x16, 16-bit): lane&15 = N col; VGPR j holds
//   K = (lane>>4)*16 + j*2, +1
// ---------------------------------------------------------------------------
__global__ __launch_bounds__(128) void gemm_wmma_kernel(const _Float16* __restrict__ Xh,
                                                        const _Float16* __restrict__ Wt,
                                                        float* __restrict__ H,
                                                        int Mrows, int Nc, int K) {
  const int lane = threadIdx.x;        // 0..31
  const int half = lane >> 4;
  const int l    = lane & 15;
  const int rowBase  = blockIdx.x * 16;
  const int colBase0 = (blockIdx.y * blockDim.y + threadIdx.y) * 64;  // 4 tiles
  if (colBase0 >= Nc) return;

  int m = rowBase + l;
  if (m >= Mrows) m = Mrows - 1;       // clamp loads; store guarded below

  const unsigned* __restrict__ arow = (const unsigned*)(Xh + (size_t)m * K);
  const unsigned* __restrict__ brow0 = (const unsigned*)(Wt + (size_t)(colBase0 +  0 + l) * K);
  const unsigned* __restrict__ brow1 = (const unsigned*)(Wt + (size_t)(colBase0 + 16 + l) * K);
  const unsigned* __restrict__ brow2 = (const unsigned*)(Wt + (size_t)(colBase0 + 32 + l) * K);
  const unsigned* __restrict__ brow3 = (const unsigned*)(Wt + (size_t)(colBase0 + 48 + l) * K);

  v8f c0 = {}, c1 = {}, c2 = {}, c3 = {};
  for (int k0 = 0; k0 < K; k0 += 32) {
    union { unsigned u[8]; v16h v; } A, B0, B1, B2, B3;
#pragma unroll
    for (int j = 0; j < 8; ++j) {
      const int ka = (k0 + half * 16 + j * 2) >> 1;
      const int kb = (k0 + half * 16 + j * 2) >> 1;
      A.u[j]  = arow[ka];
      B0.u[j] = brow0[kb];
      B1.u[j] = brow1[kb];
      B2.u[j] = brow2[kb];
      B3.u[j] = brow3[kb];
    }
    if (k0 + 32 < K) {
      // pull next A K-panel toward the caches while the 4 WMMAs execute
      __builtin_prefetch(&arow[(k0 + 32) >> 1], 0, 0);
    }
    c0 = __builtin_amdgcn_wmma_f32_16x16x32_f16(false, A.v, false, B0.v, (short)0, c0, false, false);
    c1 = __builtin_amdgcn_wmma_f32_16x16x32_f16(false, A.v, false, B1.v, (short)0, c1, false, false);
    c2 = __builtin_amdgcn_wmma_f32_16x16x32_f16(false, A.v, false, B2.v, (short)0, c2, false, false);
    c3 = __builtin_amdgcn_wmma_f32_16x16x32_f16(false, A.v, false, B3.v, (short)0, c3, false, false);
  }

  // C/D layout: VGPR r, lanes 0-15 -> M=r, lanes 16-31 -> M=r+8; N = lane&15
  const int mOut = rowBase + half * 8;
#pragma unroll
  for (int r = 0; r < 8; ++r) {
    const int mr = mOut + r;
    if (mr < Mrows) {
      float* hrow = H + (size_t)mr * Nc + colBase0 + l;
      hrow[0]  = c0[r];
      hrow[16] = c1[r];
      hrow[32] = c2[r];
      hrow[48] = c3[r];
    }
  }
}

// ---------------------------------------------------------------------------
// GCN edge scatter:  Macc[dst[e], :] += H[src[e], :]
// One thread per (edge, 4-float group); coalesced gather, f32 atomics scatter.
// ---------------------------------------------------------------------------
__global__ __launch_bounds__(256) void gcn_scatter_kernel(const float4* __restrict__ H4,
                                                          const int* __restrict__ src,
                                                          const int* __restrict__ dst,
                                                          float* __restrict__ Macc,
                                                          long long total, int Nc4, int Nc) {
  long long id = (long long)blockIdx.x * blockDim.x + threadIdx.x;
  long long stride = (long long)gridDim.x * blockDim.x;
  for (; id < total; id += stride) {
    int e = (int)(id / Nc4);
    int g = (int)(id % Nc4);
    float4 v = H4[(size_t)src[e] * Nc4 + g];
    float* mp = Macc + (size_t)dst[e] * Nc + g * 4;
    atomicAdd(mp + 0, v.x);
    atomicAdd(mp + 1, v.y);
    atomicAdd(mp + 2, v.z);
    atomicAdd(mp + 3, v.w);
  }
}

// out[n,o] = relu(Macc[n,o] * rsIn[n] + b[o])
__global__ __launch_bounds__(256) void gcn_finalize_kernel(const float* __restrict__ Macc,
                                                           const float* __restrict__ rsIn,
                                                           const float* __restrict__ b,
                                                           float* __restrict__ out,
                                                           int N, int O) {
  long long id = (long long)blockIdx.x * blockDim.x + threadIdx.x;
  long long total = (long long)N * O;
  if (id >= total) return;
  int n = (int)(id / O);
  int o = (int)(id % O);
  float v = Macc[id] * rsIn[n] + b[o];
  out[id] = v > 0.0f ? v : 0.0f;
}

// ---------------------------------------------------------------------------
// GNA (graph attention) path  — tiny dims, templated for full unrolling
// ---------------------------------------------------------------------------
__device__ __forceinline__ unsigned f2ord(float f) {
  unsigned u = __float_as_uint(f);
  return (u & 0x80000000u) ? ~u : (u | 0x80000000u);
}
__device__ __forceinline__ float ord2f(unsigned u) {
  return (u & 0x80000000u) ? __uint_as_float(u & 0x7fffffffu) : __uint_as_float(~u);
}

// per node: s = x@W2^T + b2 ; t1 = x@W1^T + b1 ; p = sum(a*s)
template <int DI, int DO>
__global__ __launch_bounds__(256) void gna_node_kernel(const float* __restrict__ x,
                                                       const float* __restrict__ W1, const float* __restrict__ b1,
                                                       const float* __restrict__ W2, const float* __restrict__ b2,
                                                       const float* __restrict__ a,
                                                       float* __restrict__ s, float* __restrict__ t1,
                                                       float* __restrict__ p, int N) {
  int n = blockIdx.x * blockDim.x + threadIdx.x;
  if (n >= N) return;
  float xv[DI];
#pragma unroll
  for (int i = 0; i < DI; ++i) xv[i] = x[(size_t)n * DI + i];
  float pacc = 0.0f;
#pragma unroll
  for (int o = 0; o < DO; ++o) {
    float sv = b2[o];
    float tv = b1[o];
    const float* w2r = W2 + (size_t)o * DI;
    const float* w1r = W1 + (size_t)o * DI;
#pragma unroll
    for (int i = 0; i < DI; ++i) {
      sv = fmaf(w2r[i], xv[i], sv);
      tv = fmaf(w1r[i], xv[i], tv);
    }
    s[(size_t)n * DO + o]  = sv;
    t1[(size_t)n * DO + o] = tv;
    pacc = fmaf(a[o], sv, pacc);
  }
  p[n] = pacc;
}

__global__ void dot_small_kernel(const float* __restrict__ a, const float* __restrict__ b,
                                 float* __restrict__ out, int n) {
  if (blockIdx.x == 0 && threadIdx.x == 0) {
    float s = 0.0f;
    for (int i = 0; i < n; ++i) s = fmaf(a[i], b[i], s);
    *out = s;
  }
}

// aij = p[dst]-p[src]+cb ; segment-max via ordered-uint atomicMax
__global__ __launch_bounds__(256) void gna_edge_score_kernel(const float* __restrict__ p,
                                                             const int* __restrict__ src,
                                                             const int* __restrict__ dst,
                                                             const float* __restrict__ cb,
                                                             float* __restrict__ aij,
                                                             unsigned* __restrict__ mx, int E) {
  int e = blockIdx.x * blockDim.x + threadIdx.x;
  if (e >= E) return;
  float v = p[dst[e]] - p[src[e]] + *cb;
  aij[e] = v;
  atomicMax(&mx[dst[e]], f2ord(v));
}

// ex = exp(aij - mx[dst]) ; denom[dst] += ex
__global__ __launch_bounds__(256) void gna_edge_exp_kernel(const float* __restrict__ aij,
                                                           const int* __restrict__ dst,
                                                           const unsigned* __restrict__ mx,
                                                           float* __restrict__ ex,
                                                           float* __restrict__ denom, int E) {
  int e = blockIdx.x * blockDim.x + threadIdx.x;
  if (e >= E) return;
  int d = dst[e];
  float v = __expf(aij[e] - ord2f(mx[d]));
  ex[e] = v;
  atomicAdd(&denom[d], v);
}

// res[dst,:] += ex[e] * s[src,:]
template <int DO>
__global__ __launch_bounds__(256) void gna_edge_res_kernel(const float* __restrict__ ex,
                                                           const float* __restrict__ s,
                                                           const int* __restrict__ src,
                                                           const int* __restrict__ dst,
                                                           float* __restrict__ res, int E) {
  int e = blockIdx.x * blockDim.x + threadIdx.x;
  if (e >= E) return;
  float w = ex[e];
  const float* sr = s + (size_t)src[e] * DO;
  float* rr = res + (size_t)dst[e] * DO;
#pragma unroll
  for (int o = 0; o < DO; ++o) atomicAdd(&rr[o], w * sr[o]);
}

// out = relu(res/denom' + t1)
__global__ __launch_bounds__(256) void gna_finalize_kernel(const float* __restrict__ res,
                                                           const float* __restrict__ denom,
                                                           const float* __restrict__ t1,
                                                           float* __restrict__ out,
                                                           int N, int DO) {
  long long id = (long long)blockIdx.x * blockDim.x + threadIdx.x;
  long long total = (long long)N * DO;
  if (id >= total) return;
  int n = (int)(id / DO);
  float d = denom[n];
  d = (d == 0.0f) ? 1.0f : d;
  float v = res[id] / d + t1[id];
  out[id] = v > 0.0f ? v : 0.0f;
}

// ---------------------------------------------------------------------------
// Host driver
// ---------------------------------------------------------------------------
static inline unsigned nblk(long long n, int b) { return (unsigned)((n + b - 1) / b); }

extern "C" void kernel_launch(void* const* d_in, const int* in_sizes, int n_in,
                              void* d_out, int out_size, void* d_ws, size_t ws_size,
                              hipStream_t stream) {
  (void)n_in; (void)out_size; (void)ws_size;

  const int ATTR_D = 512;
  const int N = in_sizes[0] / ATTR_D;   // 50000
  const int E = in_sizes[2];            // 1600000

  const float* attr_in   = (const float*)d_in[0];
  const float* struct_in = (const float*)d_in[1];
  const int*   src       = (const int*)d_in[2];
  const int*   dst       = (const int*)d_in[3];

  // --- workspace carve-up ---
  char* ws = (char*)d_ws;
  size_t off = 0;
  auto take = [&](size_t bytes) -> void* {
    void* p = ws + off;
    off = (off + bytes + 255) & ~(size_t)255;
    return p;
  };
  float*     rsOut  = (float*)take((size_t)N * 4);
  float*     rsIn   = (float*)take((size_t)N * 4);
  float*     Xbuf   = (float*)take((size_t)N * 512 * 4);   // features / scatter accumulator
  float*     Hbuf   = (float*)take((size_t)N * 512 * 4);   // GEMM output (scatter source)
  _Float16*  Xh     = (_Float16*)take((size_t)N * 512 * 2);
  _Float16*  Wt     = (_Float16*)take((size_t)512 * 512 * 2);
  float*     S0     = (float*)take((size_t)N * 32 * 4);
  float*     S1     = (float*)take((size_t)N * 32 * 4);
  float*     sArr   = (float*)take((size_t)N * 32 * 4);
  float*     tArr   = (float*)take((size_t)N * 32 * 4);
  float*     resArr = (float*)take((size_t)N * 32 * 4);
  float*     pArr   = (float*)take((size_t)N * 4);
  unsigned*  mxArr  = (unsigned*)take((size_t)N * 4);
  float*     denArr = (float*)take((size_t)N * 4);
  float*     cbPtr  = (float*)take(256);
  float*     aijArr = (float*)take((size_t)E * 4);
  float*     exArr  = (float*)take((size_t)E * 4);

  // --- degrees (once; fixed across layers) ---
  zero_f32_kernel<<<nblk(N, 256), 256, 0, stream>>>(rsOut, N);
  zero_f32_kernel<<<nblk(N, 256), 256, 0, stream>>>(rsIn, N);
  degree_kernel<<<nblk(E, 256), 256, 0, stream>>>(src, dst, rsOut, rsIn, E);
  rsqrt_clip_kernel<<<nblk(N, 256), 256, 0, stream>>>(rsOut, N);
  rsqrt_clip_kernel<<<nblk(N, 256), 256, 0, stream>>>(rsIn, N);

  // --- GCN stack (WMMA GEMM + edge scatter) ---
  const int Kd[4] = {512, 256, 256, 256};
  const int Od[4] = {256, 256, 256, 512};
  for (int i = 0; i < 4; ++i) {
    const float* Xsrc = (i == 0) ? attr_in : Xbuf;
    const float* W = (const float*)d_in[4 + 2 * i];
    const float* b = (const float*)d_in[5 + 2 * i];
    const int K = Kd[i], O = Od[i];

    convert_scale_f16_kernel<<<nblk((long long)N * K, 256), 256, 0, stream>>>(Xsrc, rsOut, Xh, N, K);
    transpose_w_f16_kernel<<<nblk((long long)K * O, 256), 256, 0, stream>>>(W, Wt, K, O);
    zero_f32_kernel<<<nblk((long long)N * O, 256), 256, 0, stream>>>(Xbuf, (long long)N * O);

    // each wave: 16 rows x 64 cols; block of 4 waves: 16 x 256 cols
    dim3 gg(nblk(N, 16), nblk(O, 256));
    gemm_wmma_kernel<<<gg, dim3(32, 4), 0, stream>>>(Xh, Wt, Hbuf, N, O, K);

    const long long total = (long long)E * (O / 4);
    unsigned sg = nblk(total, 256);
    if (sg > (1u << 19)) sg = 1u << 19;
    gcn_scatter_kernel<<<sg, 256, 0, stream>>>((const float4*)Hbuf, src, dst, Xbuf, total, O / 4, O);

    float* Xout = (i == 3) ? (float*)d_out : Xbuf;
    gcn_finalize_kernel<<<nblk((long long)N * O, 256), 256, 0, stream>>>(Xbuf, rsIn, b, Xout, N, O);
  }

  // --- GNA stack (edge softmax attention, tiny dims) ---
  const int gdi[4] = {6, 32, 32, 32};
  const int gdo[4] = {32, 32, 32, 6};
  const float* Sin = struct_in;
  for (int i = 0; i < 4; ++i) {
    const float* W1 = (const float*)d_in[12 + 5 * i];
    const float* b1 = (const float*)d_in[13 + 5 * i];
    const float* W2 = (const float*)d_in[14 + 5 * i];
    const float* b2 = (const float*)d_in[15 + 5 * i];
    const float* av = (const float*)d_in[16 + 5 * i];
    const int DI = gdi[i], DO = gdo[i];

    if (DI == 6 && DO == 32)
      gna_node_kernel<6, 32><<<nblk(N, 256), 256, 0, stream>>>(Sin, W1, b1, W2, b2, av, sArr, tArr, pArr, N);
    else if (DI == 32 && DO == 32)
      gna_node_kernel<32, 32><<<nblk(N, 256), 256, 0, stream>>>(Sin, W1, b1, W2, b2, av, sArr, tArr, pArr, N);
    else
      gna_node_kernel<32, 6><<<nblk(N, 256), 256, 0, stream>>>(Sin, W1, b1, W2, b2, av, sArr, tArr, pArr, N);

    dot_small_kernel<<<1, 32, 0, stream>>>(av, b2, cbPtr, DO);

    fill_u32_kernel<<<nblk(N, 256), 256, 0, stream>>>(mxArr, 0x007FFFFFu /* f2ord(-inf) */, N);
    zero_f32_kernel<<<nblk(N, 256), 256, 0, stream>>>(denArr, N);
    zero_f32_kernel<<<nblk((long long)N * DO, 256), 256, 0, stream>>>(resArr, (long long)N * DO);

    gna_edge_score_kernel<<<nblk(E, 256), 256, 0, stream>>>(pArr, src, dst, cbPtr, aijArr, mxArr, E);
    gna_edge_exp_kernel<<<nblk(E, 256), 256, 0, stream>>>(aijArr, dst, mxArr, exArr, denArr, E);
    if (DO == 32)
      gna_edge_res_kernel<32><<<nblk(E, 256), 256, 0, stream>>>(exArr, sArr, src, dst, resArr, E);
    else
      gna_edge_res_kernel<6><<<nblk(E, 256), 256, 0, stream>>>(exArr, sArr, src, dst, resArr, E);

    float* Sout = (i == 3) ? ((float*)d_out + (size_t)N * 512) : ((i & 1) ? S0 : S1);
    gna_finalize_kernel<<<nblk((long long)N * DO, 256), 256, 0, stream>>>(resArr, denArr, tArr, Sout, N, DO);
    Sin = Sout;
  }
}
